// FeedForward_29265907155526
// MI455X (gfx1250) — 8-bit matrix operand order repaired and checked by reading, compiled, not run on hardware
//
#include <hip/hip_runtime.h>
#include <hip/hip_bf16.h>
#include <stdint.h>

#define EPSQ   1e-5f
#define D_MODEL 2048
#define D_FFN   8192
#define NTOK    8192

typedef __attribute__((ext_vector_type(8))) int v8i;
typedef __attribute__((ext_vector_type(4))) int v4i;

__device__ __forceinline__ v8i mk8(v4i lo, v4i hi) {
    v8i r;
    r[0] = lo[0]; r[1] = lo[1]; r[2] = lo[2]; r[3] = lo[3];
    r[4] = hi[0]; r[5] = hi[1]; r[6] = hi[2]; r[7] = hi[3];
    return r;
}

__device__ __forceinline__ void stage_A_swz(const int8_t* __restrict__ Ag,
                                            int row0, int lgDpr, int* Alds, int tid) {
    const int dpr = 1 << lgDpr;
    const int total = dpr << 4;
    const int* src = (const int*)Ag;
    for (int idx = tid; idx < total; idx += 256) {
        int r    = idx >> lgDpr;
        int wd   = idx & (dpr - 1);
        int kb   = wd << 2;
        int s    = kb >> 6;
        int w4   = (kb & 63) >> 2;
        int half = (w4 >> 2) & 1;
        int v    = (w4 & 3) | ((w4 >> 3) << 2);
        Alds[s * 256 + ((r + (half << 4)) << 3) + v] =
            src[(size_t)(row0 + r) * dpr + wd];
    }
}

__global__ __launch_bounds__(256) void k_abssum_partial(const float* __restrict__ w,
                                                        int n, float* __restrict__ part) {
    __shared__ float sm[256];
    float s = 0.0f;
    for (int i = blockIdx.x * 256 + threadIdx.x; i < n; i += gridDim.x * 256)
        s += fabsf(w[i]);
    sm[threadIdx.x] = s; __syncthreads();
    for (int off = 128; off > 0; off >>= 1) {
        if (threadIdx.x < off) sm[threadIdx.x] += sm[threadIdx.x + off];
        __syncthreads();
    }
    if (threadIdx.x == 0) part[blockIdx.x] = sm[0];
}

__global__ __launch_bounds__(256) void k_abssum_final(const float* __restrict__ part,
                                                      int np, float n, float* __restrict__ scale) {
    __shared__ float sm[256];
    float s = 0.0f;
    for (int i = threadIdx.x; i < np; i += 256) s += part[i];
    sm[threadIdx.x] = s; __syncthreads();
    for (int off = 128; off > 0; off >>= 1) {
        if (threadIdx.x < off) sm[threadIdx.x] += sm[threadIdx.x + off];
        __syncthreads();
    }
    if (threadIdx.x == 0) scale[0] = fmaxf(s = sm[0] / n, EPSQ);
}

__global__ __launch_bounds__(256) void k_wquant(const float* __restrict__ w,
                                                const float* __restrict__ scale,
                                                int8_t* __restrict__ wq, int n4) {
    const float inv = 1.0f / scale[0];
    int* dst = (int*)wq;
    for (int i = blockIdx.x * 256 + threadIdx.x; i < n4; i += gridDim.x * 256) {
        int packed = 0;
#pragma unroll
        for (int j = 0; j < 4; ++j) {
            float q = rintf(w[(size_t)i * 4 + j] * inv);
            q = fminf(fmaxf(q, -1.0f), 1.0f);
            packed |= ((int)q & 0xFF) << (8 * j);
        }
        dst[i] = packed;
    }
}

__global__ __launch_bounds__(256) void k_xquant(const float* __restrict__ x,
                                                int8_t* __restrict__ xq,
                                                float* __restrict__ xdesc) {
    __shared__ float sm[256];
    const float* row = x + (size_t)blockIdx.x * D_MODEL;
    float m = 0.0f;
    for (int i = threadIdx.x; i < D_MODEL; i += 256) m = fmaxf(m, fabsf(row[i]));
    sm[threadIdx.x] = m; __syncthreads();
    for (int off = 128; off > 0; off >>= 1) {
        if (threadIdx.x < off) sm[threadIdx.x] = fmaxf(sm[threadIdx.x], sm[threadIdx.x + off]);
        __syncthreads();
    }
    const float amax = fmaxf(sm[0], EPSQ);
    const float sc = 127.0f / amax;
    int* dst = (int*)xq + (size_t)blockIdx.x * (D_MODEL / 4);
    for (int i = threadIdx.x; i < D_MODEL / 4; i += 256) {
        int packed = 0;
#pragma unroll
        for (int j = 0; j < 4; ++j) {
            float q = rintf(row[i * 4 + j] * sc);
            q = fminf(fmaxf(q, -128.0f), 127.0f);
            packed |= ((int)q & 0xFF) << (8 * j);
        }
        dst[i] = packed;
    }
    if (threadIdx.x == 0) xdesc[blockIdx.x] = amax / 127.0f;
}

__global__ __launch_bounds__(256) void k_gate_up(
    const int8_t* __restrict__ xq, const int8_t* __restrict__ wqg,
    const int8_t* __restrict__ wqu, const float* __restrict__ sg,
    const float* __restrict__ su, const float* __restrict__ xdesc,
    int8_t* __restrict__ hq, float* __restrict__ hdesc) {
    extern __shared__ char smem[];
    __hip_bfloat16* hbuf = (__hip_bfloat16*)smem;
    int*   Alds   = (int*)(smem + 262144);
    int*   rowmax = (int*)(smem + 262144 + 32768);
    float* dscL   = (float*)(smem + 262144 + 32768 + 64);

    const int tid  = threadIdx.x;
    const int row0 = blockIdx.x << 4;
    if (tid < 16) { rowmax[tid] = 0; dscL[tid] = xdesc[row0 + tid]; }
    stage_A_swz(xq, row0, 9  , Alds, tid);
    __syncthreads();

    const float wgs = sg[0], wus = su[0];
    const int wave = tid >> 5, lane = tid & 31;
    const int colL = lane & 15;
    const int rAdd = (lane & 16) ? 8 : 0;
    const int bOff0 = (lane & 16) ? 16 : 0;
    const char* Ab = (const char*)Alds;

    for (int nt = wave; nt < (D_FFN / 16); nt += 8) {
        const int col0 = nt << 4;
        const int8_t* bg = wqg + (size_t)(col0 + colL) * D_MODEL;
        const int8_t* bu = wqu + (size_t)(col0 + colL) * D_MODEL;
        v8i accg = {0, 0, 0, 0, 0, 0, 0, 0};
        v8i accu = {0, 0, 0, 0, 0, 0, 0, 0};
#pragma unroll 4
        for (int k = 0; k < (D_MODEL / 64); ++k) {
            const v4i* ap = (const v4i*)(Ab + k * 1024 + lane * 32);
            v8i a = mk8(ap[0], ap[1]);
            const int off = k * 64 + bOff0;
            v8i Bg = mk8(*(const v4i*)(bg + off), *(const v4i*)(bg + off + 32));
            v8i Bu = mk8(*(const v4i*)(bu + off), *(const v4i*)(bu + off + 32));
            accg = __builtin_amdgcn_wmma_i32_16x16x64_iu8(true, a, true, Bg, accg, false, false);
            accu = __builtin_amdgcn_wmma_i32_16x16x64_iu8(true, a, true, Bu, accu, false, false);
        }
        const int col = col0 + colL;
#pragma unroll
        for (int v = 0; v < 8; ++v) {
            const int rl = v + rAdd;
            const float xd = dscL[rl];
            const float g = (float)accg[v] * (wgs * xd);
            const float u = (float)accu[v] * (wus * xd);
            const float h = (g / (1.0f + __expf(-g))) * u;
            atomicMax(&rowmax[rl], __float_as_int(fabsf(h)));
            hbuf[(rl << 13) + col] = __float2bfloat16(h);
        }
    }
    __syncthreads();

    int* hq32 = (int*)hq;
    for (int idx = tid; idx < 16 * (D_FFN / 4); idx += 256) {
        const int row = idx >> 11;
        const int cd  = idx & 2047;
        const float amax = fmaxf(__int_as_float(rowmax[row]), EPSQ);
        const float sc = 127.0f / amax;
        int packed = 0;
#pragma unroll
        for (int j = 0; j < 4; ++j) {
            float hv = __bfloat162float(hbuf[(row << 13) + (cd << 2) + j]);
            float q = rintf(hv * sc);
            q = fminf(fmaxf(q, -128.0f), 127.0f);
            packed |= ((int)q & 0xFF) << (8 * j);
        }
        hq32[(size_t)(row0 + row) * (D_FFN / 4) + cd] = packed;
    }
    if (tid < 16)
        hdesc[row0 + tid] = fmaxf(__int_as_float(rowmax[tid]), EPSQ) / 127.0f;
}

__global__ __launch_bounds__(256) void k_down(
    const int8_t* __restrict__ hq, const int8_t* __restrict__ wqd,
    const float* __restrict__ sd, const float* __restrict__ hdesc,
    float* __restrict__ out) {
    extern __shared__ char smem[];
    int*   Alds = (int*)smem;
    float* dscL = (float*)(smem + 131072);
    const int tid  = threadIdx.x;
    const int row0 = blockIdx.x << 4;
    if (tid < 16) dscL[tid] = hdesc[row0 + tid];
    stage_A_swz(hq, row0, 11  , Alds, tid);
    __syncthreads();

    const float wds = sd[0];
    const int wave = tid >> 5, lane = tid & 31;
    const int colL = lane & 15;
    const int rAdd = (lane & 16) ? 8 : 0;
    const int bOff0 = (lane & 16) ? 16 : 0;
    const char* Ab = (const char*)Alds;

    for (int nt = wave; nt < (D_MODEL / 16); nt += 8) {
        const int col0 = nt << 4;
        const int8_t* bp = wqd + (size_t)(col0 + colL) * D_FFN;
        v8i acc = {0, 0, 0, 0, 0, 0, 0, 0};
#pragma unroll 4
        for (int k = 0; k < (D_FFN / 64); ++k) {
            const v4i* ap = (const v4i*)(Ab + k * 1024 + lane * 32);
            v8i a = mk8(ap[0], ap[1]);
            const int off = k * 64 + bOff0;
            v8i B = mk8(*(const v4i*)(bp + off), *(const v4i*)(bp + off + 32));
            acc = __builtin_amdgcn_wmma_i32_16x16x64_iu8(true, a, true, B, acc, false, false);
        }
        const int col = col0 + colL;
#pragma unroll
        for (int v = 0; v < 8; ++v) {
            const int rl = v + rAdd;
            out[(size_t)(row0 + rl) * D_MODEL + col] = (float)acc[v] * (wds * dscL[rl]);
        }
    }
}

extern "C" void kernel_launch(void* const* d_in, const int* in_sizes, int n_in,
                              void* d_out, int out_size, void* d_ws, size_t ws_size,
                              hipStream_t stream) {
    (void)in_sizes; (void)n_in; (void)out_size; (void)ws_size;
    const float* x  = (const float*)d_in[0];
    const float* wg = (const float*)d_in[1];
    const float* wu = (const float*)d_in[2];
    const float* wd = (const float*)d_in[3];
    float* out = (float*)d_out;

    char* p = (char*)d_ws;
    auto take = [&](size_t b) -> char* {
        char* r = p;
        p += (b + 255) & ~(size_t)255;
        return r;
    };
    int8_t* wqg = (int8_t*)take((size_t)D_FFN * D_MODEL);
    int8_t* wqu = (int8_t*)take((size_t)D_FFN * D_MODEL);
    int8_t* wqd = (int8_t*)take((size_t)D_MODEL * D_FFN);
    int8_t* xq  = (int8_t*)take((size_t)NTOK * D_MODEL);
    int8_t* hq  = (int8_t*)take((size_t)NTOK * D_FFN);
    float* xdesc  = (float*)take((size_t)NTOK * 4);
    float* hdesc  = (float*)take((size_t)NTOK * 4);
    float* scales = (float*)take(256);
    float* parts  = (float*)take(3 * 512 * 4);

    (void)hipFuncSetAttribute((const void*)k_gate_up,
                              hipFuncAttributeMaxDynamicSharedMemorySize, 262144 + 32768 + 128);
    (void)hipFuncSetAttribute((const void*)k_down,
                              hipFuncAttributeMaxDynamicSharedMemorySize, 131072 + 64);

    const int NW = D_FFN * D_MODEL;
    k_abssum_partial<<<512, 256, 0, stream>>>(wg, NW, parts);
    k_abssum_partial<<<512, 256, 0, stream>>>(wu, NW, parts + 512);
    k_abssum_partial<<<512, 256, 0, stream>>>(wd, NW, parts + 1024);
    k_abssum_final<<<1, 256, 0, stream>>>(parts,        512, (float)NW, scales + 0);
    k_abssum_final<<<1, 256, 0, stream>>>(parts + 512,  512, (float)NW, scales + 1);
    k_abssum_final<<<1, 256, 0, stream>>>(parts + 1024, 512, (float)NW, scales + 2);
    k_wquant<<<2048, 256, 0, stream>>>(wg, scales + 0, wqg, NW / 4);
    k_wquant<<<2048, 256, 0, stream>>>(wu, scales + 1, wqu, NW / 4);
    k_wquant<<<2048, 256, 0, stream>>>(wd, scales + 2, wqd, NW / 4);
    k_xquant<<<NTOK, 256, 0, stream>>>(x, xq, xdesc);
    k_gate_up<<<NTOK / 16, 256, 262144 + 32768 + 128, stream>>>(
        xq, wqg, wqu, scales + 0, scales + 1, xdesc, hq, hdesc);
    k_down<<<NTOK / 16, 256, 131072 + 64, stream>>>(hq, wqd, scales + 2, hdesc, out);
}
